// MixtralMoE_16733192585651
// MI455X (gfx1250) — matrix operand order repaired and checked by reading, not compiled, not run on hardware
//
#include <hip/hip_runtime.h>
#include <hip/hip_bf16.h>

// ---------------- problem constants ----------------
#define E_      8
#define KTOP    2
#define NTOK    4096        // B*T
#define CDIM    1024
#define FDIM    3584
#define WELEMS  ((size_t)E_ * FDIM * CDIM)   // elements in w1 (== w2 == w3)

// ---------------- ffn tiling ----------------
#define MT        32        // tokens per block (two WMMA M-tiles -> 2x intensity)
#define FCHUNK    128       // F processed per chunk (8 waves x 16)
#define XS_STRIDE 1032      // 1024 + 8 pad (ushort) -> conflict-free LDS rows
#define HS_STRIDE 136       // 128 + 8 pad

typedef __attribute__((ext_vector_type(16))) __bf16 v16bf;
typedef __attribute__((ext_vector_type(8)))  float  v8f;

union V16U { v16bf v; unsigned short u[16]; uint4 q[2]; };

__device__ __forceinline__ float fast_rcp(float x) {
#if __has_builtin(__builtin_amdgcn_rcpf)
  return __builtin_amdgcn_rcpf(x);       // v_rcp_f32
#else
  return 1.0f / x;
#endif
}

__device__ __forceinline__ unsigned short f2bf(float f) {
  union { float f; unsigned u; } b; b.f = f;
  unsigned r = b.u + 0x7FFFu + ((b.u >> 16) & 1u);   // round-to-nearest-even
  return (unsigned short)(r >> 16);
}

__device__ __forceinline__ v8f wmma_bf16(v16bf a, v16bf b, v8f c) {
  // D = A(16x32 bf16) x B(32x16 bf16) + C(16x16 f32)
  return __builtin_amdgcn_wmma_f32_16x16x32_bf16(false, a, false, b,
                                                 (short)0, c, false, false);
}

// 16 consecutive f32 -> 16 bf16 (fallback path B operand)
__device__ __forceinline__ V16U load_b16_f32(const float* __restrict__ p) {
  V16U r;
  const float4* q = (const float4*)p;
#pragma unroll
  for (int i = 0; i < 4; ++i) {
    float4 f = q[i];
    r.u[4*i+0] = f2bf(f.x);
    r.u[4*i+1] = f2bf(f.y);
    r.u[4*i+2] = f2bf(f.z);
    r.u[4*i+3] = f2bf(f.w);
  }
  return r;
}

// 16 consecutive bf16 from pre-converted weights (fast path: 2x b128 loads)
__device__ __forceinline__ V16U load_b16_pre(const unsigned short* __restrict__ p) {
  V16U r;
  r.q[0] = *(const uint4*)p;
  r.q[1] = *(const uint4*)(p + 8);
  return r;
}

__device__ __forceinline__ v8f vzero8() {
  v8f z;
#pragma unroll
  for (int i = 0; i < 8; ++i) z[i] = 0.0f;
  return z;
}

// ---------------- kernel 0: zero counters ----------------
__global__ void moe_init_kernel(int* cnt, float* psum) {
  int t = threadIdx.x;
  if (t < E_) { cnt[t] = 0; psum[t] = 0.0f; }
}

// ---------------- weight f32 -> bf16 pre-convert (8 elems / thread) ----------------
__global__ void moe_cvt_bf16_kernel(const float* __restrict__ src,
                                    unsigned short* __restrict__ dst, int n8) {
  int i = blockIdx.x * 256 + threadIdx.x;
  if (i >= n8) return;
  size_t base = (size_t)i * 8;
  float4 a = *(const float4*)(src + base);
  float4 b = *(const float4*)(src + base + 4);
  union { unsigned short u[8]; uint4 q; } o;
  o.u[0] = f2bf(a.x); o.u[1] = f2bf(a.y); o.u[2] = f2bf(a.z); o.u[3] = f2bf(a.w);
  o.u[4] = f2bf(b.x); o.u[5] = f2bf(b.y); o.u[6] = f2bf(b.z); o.u[7] = f2bf(b.w);
  *(uint4*)(dst + base) = o.q;
}

// ---------------- kernel 1: router (one wave32 per token) ----------------
__global__ void moe_router_kernel(const float* __restrict__ x,
                                  const float* __restrict__ rw,
                                  int* __restrict__ cnt,
                                  float* __restrict__ psum,
                                  int* __restrict__ toks,
                                  float* __restrict__ gates,
                                  int* __restrict__ slots) {
  int wave = threadIdx.x >> 5;
  int lane = threadIdx.x & 31;
  int t = blockIdx.x * 8 + wave;
  if (t >= NTOK) return;
  const float* xt = x + (size_t)t * CDIM;

  float acc[E_];
#pragma unroll
  for (int e = 0; e < E_; ++e) acc[e] = 0.0f;

  for (int c = lane; c < CDIM; c += 32) {
    float xv = xt[c];
#pragma unroll
    for (int e = 0; e < E_; ++e) acc[e] += xv * rw[e * CDIM + c];
  }
#pragma unroll
  for (int e = 0; e < E_; ++e)
#pragma unroll
    for (int off = 16; off > 0; off >>= 1)
      acc[e] += __shfl_xor(acc[e], off, 32);

  if (lane == 0) {
    float mx = acc[0];
#pragma unroll
    for (int e = 1; e < E_; ++e) mx = fmaxf(mx, acc[e]);
    float ex[E_], s = 0.0f;
#pragma unroll
    for (int e = 0; e < E_; ++e) { ex[e] = __expf(acc[e] - mx); s += ex[e]; }
    float inv = fast_rcp(s);
#pragma unroll
    for (int e = 0; e < E_; ++e) atomicAdd(&psum[e], ex[e] * inv);

    int i0 = 0;
#pragma unroll
    for (int e = 1; e < E_; ++e) if (ex[e] > ex[i0]) i0 = e;
    int i1 = (i0 == 0) ? 1 : 0;
#pragma unroll
    for (int e = 0; e < E_; ++e) if (e != i0 && ex[e] > ex[i1]) i1 = e;

    float p0 = ex[i0], p1 = ex[i1];
    float rs = fast_rcp(p0 + p1);
    float g0 = p0 * rs, g1 = p1 * rs;

    int pos0 = atomicAdd(&cnt[i0], 1);
    toks[i0 * NTOK + pos0] = t; gates[i0 * NTOK + pos0] = g0; slots[i0 * NTOK + pos0] = 0;
    int pos1 = atomicAdd(&cnt[i1], 1);
    toks[i1 * NTOK + pos1] = t; gates[i1 * NTOK + pos1] = g1; slots[i1 * NTOK + pos1] = 1;
  }
}

// ---------------- kernel 2: gathered expert FFN (WMMA bf16, 32-token tiles) ----------------
// PRE=true : weights pre-converted to bf16 in workspace (pure b128 -> wmma)
// PRE=false: convert f32 weights to bf16 on the fly
template <bool PRE>
__global__ __launch_bounds__(256)
void moe_ffn_kernel(const float* __restrict__ x,
                    const float* __restrict__ w1f,
                    const float* __restrict__ w2f,
                    const float* __restrict__ w3f,
                    const unsigned short* __restrict__ w1b,
                    const unsigned short* __restrict__ w2b,
                    const unsigned short* __restrict__ w3b,
                    const int* __restrict__ cnt,
                    const int* __restrict__ toks,
                    const float* __restrict__ gates,
                    const int* __restrict__ slots,
                    float* __restrict__ part /* [KTOP][NTOK][CDIM] */) {
  const int e   = blockIdx.y;
  const int m0  = blockIdx.x * MT;
  const int ce  = cnt[e];
  if (m0 >= ce) return;                    // uniform early-exit
  const int nvalid = min(MT, ce - m0);

  __shared__ unsigned short Xs[MT][XS_STRIDE];   // 66.0 KB
  __shared__ unsigned short Hs[MT][HS_STRIDE];   //  8.7 KB
  __shared__ int   sTok[MT];
  __shared__ float sGate[MT];
  __shared__ int   sSlot[MT];

  const int tid = threadIdx.x;
  if (tid < MT) {
    int idx = m0 + min(tid, nvalid - 1);   // clamp pad rows to last valid
    sTok[tid]  = toks [e * NTOK + idx];
    sGate[tid] = gates[e * NTOK + idx];
    sSlot[tid] = slots[e * NTOK + idx];
  }
  __syncthreads();

  // stage X tile (MT x CDIM) into LDS as bf16
  for (int v = tid; v < MT * (CDIM / 4); v += 256) {
    int row = v >> 8;                      // CDIM/4 == 256
    int c4  = (v & 255) << 2;
    float4 f = *(const float4*)(x + (size_t)sTok[row] * CDIM + c4);
    unsigned short* p = &Xs[row][c4];
    p[0] = f2bf(f.x); p[1] = f2bf(f.y); p[2] = f2bf(f.z); p[3] = f2bf(f.w);
  }
  __syncthreads();

  const int wv    = tid >> 5;
  const int lane  = tid & 31;
  const int lhalf = lane >> 4;             // 0 or 1
  const int ln    = lane & 15;

  const size_t eoff = (size_t)e * FDIM * CDIM;

  // wave owns C range [wv*128, wv*128+127] for BOTH 16-row M-halves
  v8f yL[8], yH[8];
#pragma unroll
  for (int nt = 0; nt < 8; ++nt) { yL[nt] = vzero8(); yH[nt] = vzero8(); }

  for (int fc = 0; fc < FDIM; fc += FCHUNK) {
    const int f = fc + wv * 16 + ln;       // this lane's B column in GEMM1
    const size_t rowoff = eoff + (size_t)f * CDIM + lhalf * 16;

    if (fc + FCHUNK < FDIM) {              // pull next chunk's rows toward L2/L0
      if constexpr (PRE) {
        __builtin_prefetch(w1b + rowoff + (size_t)FCHUNK * CDIM, 0, 1);
        __builtin_prefetch(w2b + rowoff + (size_t)FCHUNK * CDIM, 0, 1);
      } else {
        __builtin_prefetch(w1f + rowoff + (size_t)FCHUNK * CDIM, 0, 1);
        __builtin_prefetch(w2f + rowoff + (size_t)FCHUNK * CDIM, 0, 1);
      }
    }

    // GEMM1: 16x16 tiles of X@w1^T and X@w2^T for both M-halves (K = CDIM)
    v8f a1L = vzero8(), a1H = vzero8(), a2L = vzero8(), a2H = vzero8();
    for (int k0 = 0; k0 < CDIM; k0 += 32) {
      V16U AL, AH;
      AL.q[0] = *(const uint4*)&Xs[ln][k0 + lhalf * 16];
      AL.q[1] = *(const uint4*)&Xs[ln][k0 + lhalf * 16 + 8];
      AH.q[0] = *(const uint4*)&Xs[16 + ln][k0 + lhalf * 16];
      AH.q[1] = *(const uint4*)&Xs[16 + ln][k0 + lhalf * 16 + 8];
      V16U B1, B2;
      if constexpr (PRE) {
        B1 = load_b16_pre(w1b + rowoff + k0);
        B2 = load_b16_pre(w2b + rowoff + k0);
      } else {
        B1 = load_b16_f32(w1f + rowoff + k0);
        B2 = load_b16_f32(w2f + rowoff + k0);
      }
      a1L = wmma_bf16(AL.v, B1.v, a1L);    // each B operand feeds 2 wmma
      a1H = wmma_bf16(AH.v, B1.v, a1H);
      a2L = wmma_bf16(AL.v, B2.v, a2L);
      a2H = wmma_bf16(AH.v, B2.v, a2H);
    }

    // h = silu(x@w2^T) * (x@w1^T)  -> Hs (bf16, row-major)
#pragma unroll
    for (int r = 0; r < 8; ++r) {
      float sL = a2L[r];
      float gL = sL * fast_rcp(1.0f + __expf(-sL)) * a1L[r];
      Hs[r + 8 * lhalf][wv * 16 + ln] = f2bf(gL);
      float sH = a2H[r];
      float gH = sH * fast_rcp(1.0f + __expf(-sH)) * a1H[r];
      Hs[16 + r + 8 * lhalf][wv * 16 + ln] = f2bf(gH);
    }
    __syncthreads();

    // GEMM2: Y += H(32 x FCHUNK) @ w3^T slice
    for (int kf = 0; kf < FCHUNK; kf += 32) {
      V16U AL, AH;
      AL.q[0] = *(const uint4*)&Hs[ln][kf + lhalf * 16];
      AL.q[1] = *(const uint4*)&Hs[ln][kf + lhalf * 16 + 8];
      AH.q[0] = *(const uint4*)&Hs[16 + ln][kf + lhalf * 16];
      AH.q[1] = *(const uint4*)&Hs[16 + ln][kf + lhalf * 16 + 8];
#pragma unroll
      for (int nt = 0; nt < 8; ++nt) {
        int c = wv * 128 + nt * 16 + ln;   // B column (output channel)
        size_t w3off = (size_t)e * CDIM * FDIM + (size_t)c * FDIM + fc + kf + lhalf * 16;
        V16U Bv;
        if constexpr (PRE) Bv = load_b16_pre(w3b + w3off);
        else               Bv = load_b16_f32(w3f + w3off);
        yL[nt] = wmma_bf16(AL.v, Bv.v, yL[nt]);
        yH[nt] = wmma_bf16(AH.v, Bv.v, yH[nt]);
      }
    }
    __syncthreads();
  }

  // scatter gated results: each (slot, token) written exactly once -> no atomics
#pragma unroll
  for (int r = 0; r < 8; ++r) {
    int mL = r + 8 * lhalf;
    if (mL < nvalid) {
      float g = sGate[mL];
      float* dst = part + ((size_t)sSlot[mL] * NTOK + sTok[mL]) * CDIM + wv * 128 + ln;
#pragma unroll
      for (int nt = 0; nt < 8; ++nt) dst[nt * 16] = g * yL[nt][r];
    }
    int mH = 16 + r + 8 * lhalf;
    if (mH < nvalid) {
      float g = sGate[mH];
      float* dst = part + ((size_t)sSlot[mH] * NTOK + sTok[mH]) * CDIM + wv * 128 + ln;
#pragma unroll
      for (int nt = 0; nt < 8; ++nt) dst[nt * 16] = g * yH[nt][r];
    }
  }
}

// ---------------- kernel 3: combine partials ----------------
__global__ void moe_combine_kernel(const float* __restrict__ part,
                                   float* __restrict__ y) {
  size_t i = ((size_t)blockIdx.x * 256 + threadIdx.x) * 4;
  float4 a = *(const float4*)(part + i);
  float4 b = *(const float4*)(part + (size_t)NTOK * CDIM + i);
  float4 o; o.x = a.x + b.x; o.y = a.y + b.y; o.z = a.z + b.z; o.w = a.w + b.w;
  *(float4*)(y + i) = o;
}

// ---------------- kernel 4: aux loss ----------------
__global__ void moe_aux_kernel(const int* __restrict__ cnt,
                               const float* __restrict__ psum,
                               float* __restrict__ aux_out) {
  if (threadIdx.x == 0) {
    float a = 0.0f;
#pragma unroll
    for (int e = 0; e < E_; ++e) {
      float fe = (float)cnt[e] / (float)(NTOK * KTOP);
      float pe = psum[e] / (float)NTOK;
      a += fe * pe;
    }
    aux_out[0] = (float)E_ * a;
  }
}

// ---------------- host launcher ----------------
extern "C" void kernel_launch(void* const* d_in, const int* in_sizes, int n_in,
                              void* d_out, int out_size, void* d_ws, size_t ws_size,
                              hipStream_t stream) {
  const float* x  = (const float*)d_in[0];
  const float* rw = (const float*)d_in[1];
  const float* w1 = (const float*)d_in[2];
  const float* w2 = (const float*)d_in[3];
  const float* w3 = (const float*)d_in[4];
  float* y = (float*)d_out;

  char* ws = (char*)d_ws;
  int*   cnt   = (int*)  (ws);
  float* psum  = (float*)(ws + 32);
  int*   toks  = (int*)  (ws + 128);
  float* gts   = (float*)(ws + 128 + (size_t)E_ * NTOK * 4);
  int*   slts  = (int*)  (ws + 128 + (size_t)2 * E_ * NTOK * 4);
  size_t part_off = 128 + (size_t)3 * E_ * NTOK * 4 + 128;        // 128B aligned
  float* part  = (float*)(ws + part_off);
  size_t wb_off = part_off + (size_t)KTOP * NTOK * CDIM * 4;      // after partials
  wb_off = (wb_off + 255) & ~(size_t)255;
  unsigned short* w1b = (unsigned short*)(ws + wb_off);
  unsigned short* w2b = w1b + WELEMS;
  unsigned short* w3b = w2b + WELEMS;
  const size_t need_full = wb_off + 3 * WELEMS * sizeof(unsigned short);
  const bool pre = (ws_size >= need_full);   // ws_size is fixed -> deterministic

  moe_init_kernel<<<1, 32, 0, stream>>>(cnt, psum);
  moe_router_kernel<<<NTOK / 8, 256, 0, stream>>>(x, rw, cnt, psum, toks, gts, slts);

  dim3 ffn_grid(NTOK / MT, E_);
  if (pre) {
    const int n8 = (int)(WELEMS / 8);
    const int cb = (n8 + 255) / 256;
    moe_cvt_bf16_kernel<<<cb, 256, 0, stream>>>(w1, w1b, n8);
    moe_cvt_bf16_kernel<<<cb, 256, 0, stream>>>(w2, w2b, n8);
    moe_cvt_bf16_kernel<<<cb, 256, 0, stream>>>(w3, w3b, n8);
    moe_ffn_kernel<true><<<ffn_grid, 256, 0, stream>>>(
        x, w1, w2, w3, w1b, w2b, w3b, cnt, toks, gts, slts, part);
  } else {
    moe_ffn_kernel<false><<<ffn_grid, 256, 0, stream>>>(
        x, w1, w2, w3, w1b, w2b, w3b, cnt, toks, gts, slts, part);
  }

  moe_combine_kernel<<<(NTOK * CDIM / 4) / 256, 256, 0, stream>>>(part, y);
  moe_aux_kernel<<<1, 32, 0, stream>>>(cnt, psum, y + (size_t)NTOK * CDIM);
}
